// GIG_GAT_54451595378710
// MI455X (gfx1250) — matrix operand order repaired and checked by reading, not compiled, not run on hardware
//
#include <hip/hip_runtime.h>
#include <hip/hip_bf16.h>

// ---------------------------------------------------------------------------
// GIG_GAT on MI455X (gfx1250): 3x GATConv + pheno projection + 3-class head.
//
// GEMMs: activations/weights pre-converted to bf16 (one streaming pass,
// negligible at 23.3 TB/s), then v_wmma_f32_16x16x32_bf16. The shared 32x32
// A tile is staged into LDS via the gfx1250 async Global->LDS path
// (GLOBAL_LOAD_ASYNC_TO_LDS_B128 + s_wait_asynccnt), double-buffered so one
// batch is always in flight while WMMAs consume the previous one. B fragments
// are contiguous global_load_b128 from a column-major bf16 weight copy.
// Edge softmax/scatter is irregular-memory/atomic bound -> flat VMEM passes;
// the 51MB feature matrix is L2-resident (192MB L2).
// ---------------------------------------------------------------------------

typedef __attribute__((ext_vector_type(16))) __bf16 v16bf;
typedef __attribute__((ext_vector_type(8)))  __bf16 v8bf;
typedef __attribute__((ext_vector_type(8)))  float  v8f;
typedef __attribute__((ext_vector_type(4)))  int    v4i;

#define DH   128   // hidden / output feature dim of every GAT layer
#define NH   4     // heads
#define HK   32    // features per head
#define CLS  3     // output classes

#define AS1 __attribute__((address_space(1)))
#define AS3 __attribute__((address_space(3)))

#if __has_builtin(__builtin_amdgcn_global_load_async_to_lds_b128)
#define HAVE_ASYNC_LDS 1
#else
#define HAVE_ASYNC_LDS 0
#endif

#if __has_builtin(__builtin_amdgcn_s_wait_asynccnt)
#define WAIT_ASYNC(n) __builtin_amdgcn_s_wait_asynccnt(n)
#else
#define WAIT_ASYNC(n) asm volatile("s_wait_asynccnt %0" ::"n"(n) : "memory")
#endif

// ---------------- monotonic float<->uint encoding for atomicMax ------------
__device__ __forceinline__ unsigned enc_f32(float f) {
  unsigned u = __float_as_uint(f);
  return (u & 0x80000000u) ? ~u : (u | 0x80000000u);
}
__device__ __forceinline__ float dec_f32(unsigned u) {
  return (u & 0x80000000u) ? __uint_as_float(u & 0x7FFFFFFFu)
                           : __uint_as_float(~u);
}

// ---------------- zero fill -------------------------------------------------
__global__ void zero_f32(float* __restrict__ p, long n) {
  long i = (long)blockIdx.x * blockDim.x + threadIdx.x;
  if (i < n) p[i] = 0.0f;
}

// ---------------- fp32 -> bf16 streaming convert ----------------------------
__global__ void cvt_bf16(const float* __restrict__ in, __bf16* __restrict__ out,
                         long n) {
  long i = (long)blockIdx.x * blockDim.x + threadIdx.x;
  if (i < n) out[i] = (__bf16)in[i];
}

// ---------------- weight fp32 [K,N] -> bf16 column-major [N,K] --------------
__global__ void cvt_w_colmajor(const float* __restrict__ W,
                               __bf16* __restrict__ Wc, int K, int N) {
  long t = (long)blockIdx.x * blockDim.x + threadIdx.x;
  if (t >= (long)K * N) return;
  int n = (int)(t / K);
  int k = (int)(t % K);
  Wc[t] = (__bf16)W[(long)k * N + n];   // Wc[n*K + k] = W[k][n]
}

// ---------------- WMMA GEMM: C[M,N] = A[M,K] @ B[K,N] + bias ---------------
// A: bf16 row-major [M,K]. Bc: bf16 column-major [N,K]. M%32==0, K%64==0,
// N==128. Block = 256 threads = 8 waves; wave w owns columns [16w,16w+16);
// each wave computes a 32x16 tile (2 accumulators share one B fragment).
// A tile (32 rows x 64 K) staged in LDS via async loads, double-buffered.
// Fragment packing per CDNA5 ISA 7.12.2 (16-bit A 16x32 / B 32x16).
__global__ void __launch_bounds__(256)
gemm_bias_wmma_bf16(const __bf16* __restrict__ A,
                    const __bf16* __restrict__ Bc,
                    const float* __restrict__ bias,
                    float* __restrict__ C,
                    int M, int K, int N) {
  const int tid  = threadIdx.x;
  const int wave = tid >> 5;           // 0..7 -> n-tile
  const int lane = tid & 31;
  const int lo   = lane & 15;
  const int hi   = lane >> 4;          // 0 or 1
  const int m0   = blockIdx.x * 32;
  const int n0   = wave * 16;

  v8f acc0 = {}, acc1 = {};
  const __bf16* bcol = Bc + (long)(n0 + lo) * K;   // column n0+lo

#if HAVE_ASYNC_LDS
  // LDS: 2 buffers x (2 K-steps x 32 rows x 32 K) bf16 = 8KB
  __shared__ __bf16 atile[2 * 2048];

  // staging coords: 256 threads x 16B cover one 4KB batch (64 K-values)
  const int skk = tid >> 7;            // which 32-K step of the batch
  const int sr  = (tid >> 2) & 31;     // row within tile
  const int sc  = tid & 3;             // 16B chunk within 64B row
  const __bf16* gsrc = A + (long)(m0 + sr) * K + skk * 32 + sc * 8;
  const int ldst = skk * 1024 + sr * 32 + sc * 8;  // elems within one buffer

  // issue batch 0 into buffer 0
  __builtin_amdgcn_global_load_async_to_lds_b128(
      (AS1 v4i*)gsrc, (AS3 v4i*)(atile + ldst), 0, 0);

  int buf = 0;
  for (int kb = 0; kb < K; kb += 64) {
    const bool more = (kb + 64) < K;
    if (more) {   // stage next batch into the other buffer while we compute
      __builtin_amdgcn_global_load_async_to_lds_b128(
          (AS1 v4i*)(gsrc + kb + 64),
          (AS3 v4i*)(atile + (buf ^ 1) * 2048 + ldst), 0, 0);
      WAIT_ASYNC(1);                   // oldest batch (ours) complete
    } else {
      WAIT_ASYNC(0);
    }
    __syncthreads();                   // all waves' LDS writes visible

#pragma unroll
    for (int kk = 0; kk < 2; ++kk) {
      const __bf16* lt = atile + buf * 2048 + kk * 1024;  // 32x32 step tile
      // A fragments from LDS (ds_load_b128): row stride 64B
      v8bf a0lo = *(const v8bf*)(lt + lo * 32 + hi * 8);
      v8bf a0hi = *(const v8bf*)(lt + lo * 32 + 16 + hi * 8);
      v8bf a1lo = *(const v8bf*)(lt + (16 + lo) * 32 + hi * 8);
      v8bf a1hi = *(const v8bf*)(lt + (16 + lo) * 32 + 16 + hi * 8);
      v16bf a0 = __builtin_shufflevector(a0lo, a0hi, 0,1,2,3,4,5,6,7,
                                         8,9,10,11,12,13,14,15);
      v16bf a1 = __builtin_shufflevector(a1lo, a1hi, 0,1,2,3,4,5,6,7,
                                         8,9,10,11,12,13,14,15);
      // B fragment: lane (lo,hi): elems 0..7 = K[k0+hi*8 ..], elems 8..15 = K[k0+16+hi*8 ..]
      const int k0 = kb + kk * 32;
      v8bf blo = *(const v8bf*)(bcol + k0 + hi * 8);
      v8bf bhi = *(const v8bf*)(bcol + k0 + 16 + hi * 8);
      v16bf b = __builtin_shufflevector(blo, bhi, 0,1,2,3,4,5,6,7,
                                        8,9,10,11,12,13,14,15);

      acc0 = __builtin_amdgcn_wmma_f32_16x16x32_bf16(false, a0, false, b,
                                                     (short)0, acc0, false, false);
      acc1 = __builtin_amdgcn_wmma_f32_16x16x32_bf16(false, a1, false, b,
                                                     (short)0, acc1, false, false);
    }
    __syncthreads();                   // reads done before buffer reuse
    buf ^= 1;
  }
#else
  // fallback: direct global bf16 fragment loads
  const __bf16* arow0 = A + (long)(m0 + lo) * K;
  const __bf16* arow1 = A + (long)(m0 + 16 + lo) * K;
  for (int k0 = 0; k0 < K; k0 += 32) {
    v8bf a0lo = *(const v8bf*)(arow0 + k0 + hi * 8);
    v8bf a0hi = *(const v8bf*)(arow0 + k0 + 16 + hi * 8);
    v8bf a1lo = *(const v8bf*)(arow1 + k0 + hi * 8);
    v8bf a1hi = *(const v8bf*)(arow1 + k0 + 16 + hi * 8);
    v16bf a0 = __builtin_shufflevector(a0lo, a0hi, 0,1,2,3,4,5,6,7,
                                       8,9,10,11,12,13,14,15);
    v16bf a1 = __builtin_shufflevector(a1lo, a1hi, 0,1,2,3,4,5,6,7,
                                       8,9,10,11,12,13,14,15);
    v8bf blo = *(const v8bf*)(bcol + k0 + hi * 8);
    v8bf bhi = *(const v8bf*)(bcol + k0 + 16 + hi * 8);
    v16bf b = __builtin_shufflevector(blo, bhi, 0,1,2,3,4,5,6,7,
                                      8,9,10,11,12,13,14,15);
    acc0 = __builtin_amdgcn_wmma_f32_16x16x32_bf16(false, a0, false, b,
                                                   (short)0, acc0, false, false);
    acc1 = __builtin_amdgcn_wmma_f32_16x16x32_bf16(false, a1, false, b,
                                                   (short)0, acc1, false, false);
  }
#endif

  // D fragment: lane (lo,hi) -> rows m0+hi*8+v (v=0..7), col n0+lo
  const float bv = bias[n0 + lo];
  float* cp0 = C + (long)(m0 + hi * 8) * N + n0 + lo;
  float* cp1 = C + (long)(m0 + 16 + hi * 8) * N + n0 + lo;
#pragma unroll
  for (int v = 0; v < 8; ++v) {
    cp0[(long)v * N] = acc0[v] + bv;
    cp1[(long)v * N] = acc1[v] + bv;
  }
}

// ---------------- per-node attention dot products ---------------------------
// ai[n,h] = sum_k h[n,h*32+k]*att[h,0:32] ; aj[n,h] = ... att[h,32:64]
__global__ void attn_node(const float* __restrict__ h,
                          const float* __restrict__ att,
                          float* __restrict__ ai, float* __restrict__ aj,
                          int N) {
  int t = blockIdx.x * blockDim.x + threadIdx.x;
  if (t >= N * NH) return;
  int n  = t >> 2;
  int hd = t & (NH - 1);
  const float* hp = h + (long)n * DH + hd * HK;
  const float* a0 = att + hd * 2 * HK;
  float si = 0.f, sj = 0.f;
#pragma unroll
  for (int k = 0; k < HK; ++k) {
    float v = hp[k];
    si += v * a0[k];
    sj += v * a0[HK + k];
  }
  ai[t] = si;
  aj[t] = sj;
}

// ---------------- edge pass 1: alpha = leaky(ai[dst]+aj[src]); segmax(src) --
__global__ void alpha_max(const int* __restrict__ ei, long E, long Ee,
                          const float* __restrict__ ai,
                          const float* __restrict__ aj,
                          float* __restrict__ alpha,
                          unsigned* __restrict__ amax) {
  long t = (long)blockIdx.x * blockDim.x + threadIdx.x;
  if (t >= Ee * NH) return;
  long e = t >> 2;
  int hd = (int)(t & (NH - 1));
  int s, d;
  if (e < E) { s = ei[e]; d = ei[E + e]; }
  else       { s = d = (int)(e - E); }
  float a = ai[(long)d * NH + hd] + aj[(long)s * NH + hd];
  a = (a >= 0.f) ? a : 0.2f * a;   // attention leaky relu (slope 0.2)
  alpha[t] = a;
  atomicMax(&amax[(long)s * NH + hd], enc_f32(a));
}

// ---------------- edge pass 2: ex = exp(alpha - max[src]); segsum(src) ------
__global__ void exp_sum(const int* __restrict__ ei, long E, long Ee,
                        float* __restrict__ alpha,
                        const unsigned* __restrict__ amax,
                        float* __restrict__ denom) {
  long t = (long)blockIdx.x * blockDim.x + threadIdx.x;
  if (t >= Ee * NH) return;
  long e = t >> 2;
  int hd = (int)(t & (NH - 1));
  int s;
  if (e < E) s = ei[e]; else s = (int)(e - E);
  float m  = dec_f32(amax[(long)s * NH + hd]);
  float ex = __expf(alpha[t] - m);
  alpha[t] = ex;
  atomicAdd(&denom[(long)s * NH + hd], ex);
}

// ---------------- edge pass 3: acc[dst] += h[src] * ex/(denom[src]+eps) -----
__global__ void scatter_msg(const int* __restrict__ ei, long E, long Ee,
                            const float* __restrict__ h,
                            const float* __restrict__ alpha,
                            const float* __restrict__ denom,
                            float* __restrict__ acc) {
  long t = (long)blockIdx.x * blockDim.x + threadIdx.x;
  if (t >= Ee * DH) return;
  long e = t >> 7;               // DH = 128
  int  f = (int)(t & (DH - 1));
  int  hd = f >> 5;              // HK = 32
  int s, d;
  if (e < E) { s = ei[e]; d = ei[E + e]; }
  else       { s = d = (int)(e - E); }
  float w = alpha[e * NH + hd] / (denom[(long)s * NH + hd] + 1e-16f);
  atomicAdd(&acc[(long)d * DH + f], h[(long)s * DH + f] * w);
}

// ---------------- post: out = relu(acc + bias) ------------------------------
// (inter-layer leaky_relu(.,0.5) after relu is identity -> fused away)
__global__ void post_bias_relu(const float* __restrict__ acc,
                               const float* __restrict__ bias,
                               float* __restrict__ out, long n) {
  long i = (long)blockIdx.x * blockDim.x + threadIdx.x;
  if (i >= n) return;
  float v = acc[i] + bias[i & (DH - 1)];
  out[i] = (v > 0.f) ? v : 0.f;
}

// ---------------- output head: logits = g @ out_W ; ypred = argmax ----------
__global__ void out_proj(const float* __restrict__ g,
                         const float* __restrict__ W,  // [128,3] row-major
                         float* __restrict__ logits,
                         float* __restrict__ ypred, int N) {
  int n = blockIdx.x * blockDim.x + threadIdx.x;
  if (n >= N) return;
  const float* gp = g + (long)n * DH;
  float c0 = 0.f, c1 = 0.f, c2 = 0.f;
#pragma unroll 4
  for (int k = 0; k < DH; ++k) {
    float v = gp[k];
    c0 += v * W[k * CLS + 0];
    c1 += v * W[k * CLS + 1];
    c2 += v * W[k * CLS + 2];
  }
  logits[(long)n * CLS + 0] = c0;
  logits[(long)n * CLS + 1] = c1;
  logits[(long)n * CLS + 2] = c2;
  int arg = 0; float best = c0;
  if (c1 > best) { best = c1; arg = 1; }
  if (c2 > best) { best = c2; arg = 2; }
  ypred[n] = (float)arg;
}

// ---------------- node_index gathers ----------------------------------------
__global__ void gather_nodes(const int* __restrict__ idx,
                             const float* __restrict__ logits,
                             const float* __restrict__ ypred,
                             float* __restrict__ node_out,
                             float* __restrict__ y_nodepred, int NI) {
  int i = blockIdx.x * blockDim.x + threadIdx.x;
  if (i >= NI) return;
  int n = idx[i];
  node_out[(long)i * CLS + 0] = logits[(long)n * CLS + 0];
  node_out[(long)i * CLS + 1] = logits[(long)n * CLS + 1];
  node_out[(long)i * CLS + 2] = logits[(long)n * CLS + 2];
  y_nodepred[i] = ypred[n];
}

// ===========================================================================
extern "C" void kernel_launch(void* const* d_in, const int* in_sizes, int n_in,
                              void* d_out, int out_size, void* d_ws, size_t ws_size,
                              hipStream_t stream) {
  // ---- inputs (setup_inputs order) ----
  const float* x        = (const float*)d_in[0];    // [N,256]
  const int*   ei       = (const int*)  d_in[1];    // [2,E]
  const int*   node_idx = (const int*)  d_in[2];    // [NI]
  const float* pheno_W  = (const float*)d_in[3];    // [256,128]
  const float* pheno_b  = (const float*)d_in[4];    // [128]
  const float* Ws[3]    = {(const float*)d_in[5],  (const float*)d_in[9],  (const float*)d_in[13]};
  const float* bs[3]    = {(const float*)d_in[6],  (const float*)d_in[10], (const float*)d_in[14]};
  const float* atts[3]  = {(const float*)d_in[7],  (const float*)d_in[11], (const float*)d_in[15]};
  const float* biases[3]= {(const float*)d_in[8],  (const float*)d_in[12], (const float*)d_in[16]};
  const float* out_W    = (const float*)d_in[17];   // [128,3]

  const int  N   = in_sizes[0] / 256;   // 100000
  const int  KIN = in_sizes[3] / DH;    // 256
  const long E   = (long)in_sizes[1] / 2;
  const long Ee  = E + N;
  const int  NI  = in_sizes[2];

  // ---- workspace layout ----
  float* ws    = (float*)d_ws;
  long   feat  = (long)N * DH;          // 12.8M floats
  float* bufG  = ws;                    // activations (ping)
  float* bufH  = bufG + feat;           // post-GEMM features h
  float* bufAcc= bufH + feat;           // aggregation accumulator
  float* ai    = bufAcc + feat;         // [N,4]
  float* aj    = ai + (long)N * NH;
  float* amaxf = aj + (long)N * NH;     // [N,4] as uint
  float* denom = amaxf + (long)N * NH;  // [N,4]
  float* alpha = denom + (long)N * NH;  // [Ee,4]
  float* logits= alpha + Ee * NH;       // [N,3]
  __bf16* abf  = (__bf16*)(logits + (long)N * CLS);   // [N, KIN] bf16 activations
  __bf16* wcbf = abf + (long)N * KIN;                 // [128, 256] bf16 weight (col-major)
  unsigned* amax = (unsigned*)amaxf;

  // ---- d_out layout: x_embed | node_output | ypred | y_nodepred ----
  float* xe       = (float*)d_out;               // [N,128]
  float* node_out = xe + feat;                   // [NI,3]
  float* ypred    = node_out + (long)NI * CLS;   // [N]
  float* y_npred  = ypred + N;                   // [NI]

  const int BT = 256;
  dim3 blk(BT);
  auto cdiv = [](long a, long b) { return (int)((a + b - 1) / b); };

  // 1) pheno projection: g = x @ pheno_W + pheno_b  (bf16 WMMA)
  cvt_bf16<<<cdiv((long)N * KIN, BT), blk, 0, stream>>>(x, abf, (long)N * KIN);
  cvt_w_colmajor<<<cdiv((long)KIN * DH, BT), blk, 0, stream>>>(pheno_W, wcbf, KIN, DH);
  gemm_bias_wmma_bf16<<<dim3(N / 32), blk, 0, stream>>>(abf, wcbf, pheno_b, bufG,
                                                        N, KIN, DH);

  // 2) three GAT layers
  for (int l = 0; l < 3; ++l) {
    float* inPtr  = bufG;
    float* outPtr = (l < 2) ? bufG : xe;   // last layer writes x_embed directly

    // h = in @ W + b (bf16 WMMA)
    cvt_bf16<<<cdiv(feat, BT), blk, 0, stream>>>(inPtr, abf, feat);
    cvt_w_colmajor<<<cdiv((long)DH * DH, BT), blk, 0, stream>>>(Ws[l], wcbf, DH, DH);
    gemm_bias_wmma_bf16<<<dim3(N / 32), blk, 0, stream>>>(abf, wcbf, bs[l], bufH,
                                                          N, DH, DH);

    // per-node attention dots
    attn_node<<<cdiv((long)N * NH, BT), blk, 0, stream>>>(bufH, atts[l], ai, aj, N);

    // zero segmax/denom (2*N*4 floats) and accumulator
    zero_f32<<<cdiv(2L * N * NH, BT), blk, 0, stream>>>(amaxf, 2L * N * NH);
    zero_f32<<<cdiv(feat, BT), blk, 0, stream>>>(bufAcc, feat);

    // edge passes
    long tE = Ee * NH;
    alpha_max<<<cdiv(tE, BT), blk, 0, stream>>>(ei, E, Ee, ai, aj, alpha, amax);
    exp_sum  <<<cdiv(tE, BT), blk, 0, stream>>>(ei, E, Ee, alpha, amax, denom);
    long tS = Ee * DH;
    scatter_msg<<<cdiv(tS, BT), blk, 0, stream>>>(ei, E, Ee, bufH, alpha, denom, bufAcc);

    // out = relu(acc + bias)   (leaky(0.5) afterwards is identity on relu output)
    post_bias_relu<<<cdiv(feat, BT), blk, 0, stream>>>(bufAcc, biases[l], outPtr, feat);
  }

  // 3) output head + argmax
  out_proj<<<cdiv(N, BT), blk, 0, stream>>>(xe, out_W, logits, ypred, N);

  // 4) node_index gathers
  gather_nodes<<<cdiv(NI, BT), blk, 0, stream>>>(node_idx, logits, ypred, node_out, y_npred, NI);
}
